// VectorQuantizer_86620900426259
// MI455X (gfx1250) — matrix operand order repaired and checked by reading, not compiled, not run on hardware
//
#include <hip/hip_runtime.h>

typedef __attribute__((ext_vector_type(16))) __bf16 v16bf;
typedef __attribute__((ext_vector_type(8)))  float  v8f;

#define N_E   8192
#define E_DIM 256
#define N_TOK 32768

// LDS tile: 64 codebook rows, padded row stride of 264 bf16 (528 B) so that the
// 16 lanes of each ds_load_b128 map to distinct banks (528/4 = 132 ≡ 4 mod 64).
#define TILE_N    64
#define LDS_PITCH 264
#define N_TILES   (N_E / TILE_N)

// workspace layout (bytes)
#define WS_EMBB  0                        // N_E*E_DIM*2  = 4 MiB  (bf16 codebook)
#define WS_NORMS (N_E * E_DIM * 2)        // N_E*4        = 32 KiB (f32 ||e||^2)
#define WS_IDX   (WS_NORMS + N_E * 4)     // N_TOK*4      = 128 KiB (int32 argmin)
#define WS_LOSS  (WS_IDX + N_TOK * 4)     // 4 B          (f32 accumulator)

// ---------------------------------------------------------------------------
// Kernel 1: convert codebook to bf16, compute ||e||^2 per row, zero loss accum
// ---------------------------------------------------------------------------
__global__ void vq_prep(const float* __restrict__ emb, __bf16* __restrict__ embb,
                        float* __restrict__ norms, float* __restrict__ lossAccum) {
    const int lane = threadIdx.x & 31;
    const int wid  = threadIdx.x >> 5;
    const int row  = blockIdx.x * 8 + wid;
    const float* src = emb  + (size_t)row * E_DIM;
    __bf16*      dst = embb + (size_t)row * E_DIM;
    float s = 0.f;
#pragma unroll
    for (int i = 0; i < E_DIM / 32; ++i) {
        float v = src[lane + 32 * i];
        s += v * v;
        dst[lane + 32 * i] = (__bf16)v;
    }
#pragma unroll
    for (int off = 16; off >= 1; off >>= 1) s += __shfl_xor(s, off);
    if (lane == 0) norms[row] = s;
    if (blockIdx.x == 0 && threadIdx.x == 0) *lossAccum = 0.f;
}

// ---------------------------------------------------------------------------
// Kernel 2: argmin over codebook via bf16 WMMA.
// 256 blocks x 256 threads (8 waves); wave w owns tokens [blk*128+w*16, +16).
// Codebook streamed through a double-buffered LDS tile filled with
// GLOBAL_LOAD_ASYNC_TO_LDS_B128 (ASYNCcnt), overlapping copy with compute.
// ---------------------------------------------------------------------------
__global__ void __launch_bounds__(256, 2)
vq_argmin(const float* __restrict__ x, const __bf16* __restrict__ embb,
          const float* __restrict__ norms, int* __restrict__ indices) {
    __shared__ __bf16 tile[2][TILE_N * LDS_PITCH];   // 2 x 33,792 B

    const int tid  = threadIdx.x;
    const int lane = tid & 31;
    const int wid  = tid >> 5;
    const int half = lane >> 4;       // 0: lanes 0-15, 1: lanes 16-31
    const int row  = lane & 15;
    const int tokBase = blockIdx.x * 128 + wid * 16;

    // ---- A fragments: 16 tokens x 256 dims, bf16, register resident --------
    //   lane<16  (M=lane):    elems 0..15 = K 32c+0..15
    //   lane>=16 (M=lane-16): elems 0..15 = K 32c+16..31
    const float* xrow = x + (size_t)(tokBase + row) * E_DIM;
    v16bf A[8];
#pragma unroll
    for (int c = 0; c < 8; ++c) {
        const int k0 = 32 * c + (half << 4);
#pragma unroll
        for (int j = 0; j < 8; ++j) A[c][j]     = (__bf16)xrow[k0 + j];
#pragma unroll
        for (int j = 0; j < 8; ++j) A[c][8 + j] = (__bf16)xrow[k0 + 8 + j];
    }

    float bestVal[8];
    int   bestIdx[8];
#pragma unroll
    for (int r = 0; r < 8; ++r) { bestVal[r] = 3.4e38f; bestIdx[r] = 0; }

    // async-copy assignment: thread -> (row lr, 128-byte quarter lq)
    const int lr = tid >> 2;
    const int lq = tid & 3;
    const unsigned long long gBase = (unsigned long long)embb;
    const unsigned gOffThread = (unsigned)((lr * E_DIM + lq * 64) * 2); // bytes
    const unsigned ldsThread0 = (unsigned)(size_t)(&tile[0][lr * LDS_PITCH + lq * 64]);
    const unsigned ldsThread1 = (unsigned)(size_t)(&tile[1][lr * LDS_PITCH + lq * 64]);

    // prologue: kick off tile 0 into buffer 0
    {
        const unsigned gOff = gOffThread;   // tile 0
#pragma unroll
        for (int i = 0; i < 8; ++i) {
            asm volatile("global_load_async_to_lds_b128 %0, %1, %2"
                         :: "v"(ldsThread0 + i * 16), "v"(gOff + i * 16), "s"(gBase)
                         : "memory");
        }
    }

    for (int it = 0; it < N_TILES; ++it) {
        const int buf = it & 1;
        // my async loads for this tile have landed...
        asm volatile("s_wait_asynccnt 0" ::: "memory");
        // ...and everyone is past the compute that used the other buffer
        __syncthreads();

        if (it + 1 < N_TILES) {   // kick off next tile into the other buffer
            const unsigned gOff = gOffThread + (unsigned)((it + 1) * TILE_N * E_DIM * 2);
            const unsigned ldsDst = (buf == 0) ? ldsThread1 : ldsThread0;
#pragma unroll
            for (int i = 0; i < 8; ++i) {
                asm volatile("global_load_async_to_lds_b128 %0, %1, %2"
                             :: "v"(ldsDst + i * 16), "v"(gOff + i * 16), "s"(gBase)
                             : "memory");
            }
        }

        // ---- compute on tile[buf]: 4 N-subtiles x 8 K-chunks ---------------
        // B fragments, same k order as A (not ISA table order): lane holds column N=(lane&15);
        // lanes 0-15: K 32c+0..15, lanes 16-31: K 32c+16..31 (contiguous in row)
        const __bf16* base = tile[buf];
        const __bf16* brow[4];
#pragma unroll
        for (int s = 0; s < 4; ++s)
            brow[s] = base + (s * 16 + row) * LDS_PITCH + (half << 4);

        v8f acc[4];
#pragma unroll
        for (int s = 0; s < 4; ++s) acc[s] = (v8f){0.f,0.f,0.f,0.f,0.f,0.f,0.f,0.f};

        v16bf B[4], Bn[4];
#pragma unroll
        for (int s = 0; s < 4; ++s) B[s] = *(const v16bf*)(brow[s]);

#pragma unroll
        for (int c = 0; c < 8; ++c) {
            if (c < 7) {          // prefetch next K-chunk for all 4 subtiles
#pragma unroll
                for (int s = 0; s < 4; ++s)
                    Bn[s] = *(const v16bf*)(brow[s] + 32 * (c + 1));
            }
#pragma unroll
            for (int s = 0; s < 4; ++s)
                acc[s] = __builtin_amdgcn_wmma_f32_16x16x32_bf16(
                            false, A[c], false, B[s], (short)0, acc[s], false, false);
#pragma unroll
            for (int s = 0; s < 4; ++s) B[s] = Bn[s];
        }

        const int t0 = it * TILE_N;
#pragma unroll
        for (int s = 0; s < 4; ++s) {
            const int n = t0 + s * 16 + row;       // codebook column this lane owns
            const float nrm = norms[n];
            // C/D layout: acc[s][r] is (M = r + 8*half, N = row)
#pragma unroll
            for (int r = 0; r < 8; ++r) {
                const float d = nrm - 2.0f * acc[s][r];
                if (d < bestVal[r]) { bestVal[r] = d; bestIdx[r] = n; }
            }
        }
    }

    // reduce across the 16 lanes of each half (xor masks 1..8 stay in-half)
#pragma unroll
    for (int r = 0; r < 8; ++r) {
        float v  = bestVal[r];
        int   ix = bestIdx[r];
#pragma unroll
        for (int off = 8; off >= 1; off >>= 1) {
            float ov = __shfl_xor(v, off);
            int   oi = __shfl_xor(ix, off);
            if (ov < v || (ov == v && oi < ix)) { v = ov; ix = oi; }
        }
        if (row == 0) indices[tokBase + r + (half << 3)] = ix;   // M = r + 8*half
    }
}

// ---------------------------------------------------------------------------
// Kernel 3: x_q = emb[idx] (f32), accumulate sum((x_q - x)^2)
// ---------------------------------------------------------------------------
__global__ void vq_gather(const float* __restrict__ x, const float* __restrict__ emb,
                          const int* __restrict__ indices, float* __restrict__ xq,
                          float* __restrict__ lossAccum) {
    __shared__ float wsum[8];
    const int t = blockIdx.x;
    const int d = threadIdx.x;
    const int id = indices[t];
    const float e  = emb[(size_t)id * E_DIM + d];
    const float xv = x[(size_t)t * E_DIM + d];
    xq[(size_t)t * E_DIM + d] = e;
    float s = (e - xv) * (e - xv);
#pragma unroll
    for (int off = 16; off >= 1; off >>= 1) s += __shfl_xor(s, off);
    if ((threadIdx.x & 31) == 0) wsum[threadIdx.x >> 5] = s;
    __syncthreads();
    if (threadIdx.x == 0) {
        float b = 0.f;
#pragma unroll
        for (int i = 0; i < 8; ++i) b += wsum[i];
        atomicAdd(lossAccum, b);
    }
}

// ---------------------------------------------------------------------------
// Kernel 4: loss = (1 + MU) * sum / (N_TOK*E_DIM); indices -> float tail
// ---------------------------------------------------------------------------
__global__ void vq_finalize(const int* __restrict__ indices,
                            const float* __restrict__ lossAccum,
                            float* __restrict__ outLoss, float* __restrict__ outIdx) {
    const int i = blockIdx.x * 256 + threadIdx.x;
    outIdx[i] = (float)indices[i];
    if (i == 0)
        *outLoss = 1.25f * (*lossAccum) / ((float)N_TOK * (float)E_DIM);
}

// ---------------------------------------------------------------------------
extern "C" void kernel_launch(void* const* d_in, const int* in_sizes, int n_in,
                              void* d_out, int out_size, void* d_ws, size_t ws_size,
                              hipStream_t stream) {
    const float* x   = (const float*)d_in[0];   // [N_TOK, E_DIM]
    const float* emb = (const float*)d_in[1];   // [N_E, E_DIM]
    // d_in[2] = label, d_in[3] = idx : unused by the computation

    char* ws = (char*)d_ws;
    __bf16* embb    = (__bf16*)(ws + WS_EMBB);
    float*  norms   = (float*) (ws + WS_NORMS);
    int*    indices = (int*)   (ws + WS_IDX);
    float*  lossAcc = (float*) (ws + WS_LOSS);

    float* out     = (float*)d_out;
    float* xq      = out;                               // N_TOK*E_DIM
    float* outLoss = out + (size_t)N_TOK * E_DIM;       // 1
    float* outIdx  = outLoss + 1;                       // N_TOK

    vq_prep<<<N_E / 8, 256, 0, stream>>>(emb, embb, norms, lossAcc);
    vq_argmin<<<N_TOK / 128, 256, 0, stream>>>(x, embb, norms, indices);
    vq_gather<<<N_TOK, 256, 0, stream>>>(x, emb, indices, xq, lossAcc);
    vq_finalize<<<N_TOK / 256, 256, 0, stream>>>(indices, lossAcc, outLoss, outIdx);
}
